// EquivariantAttention_33852932227753
// MI455X (gfx1250) — matrix operand order repaired and checked by reading, not compiled, not run on hardware
//
#include <hip/hip_runtime.h>
#include <math.h>

// ---------------- problem constants ----------------
constexpr int NH  = 8;     // heads
constexpr int DHD = 64;    // head dim
constexpr int KNN = 32;    // neighbors
constexpr int PBD = 256;   // dpb hidden
constexpr int DEE = 16;    // edge dim
constexpr int NPT = 2048;  // nodes
constexpr int DD  = 512;   // feature dim
#define ATT_SCALE 0.125f   // DH^-0.5

// ---------------- bf16 / WMMA plumbing ----------------
typedef __bf16 bf16;
typedef __attribute__((ext_vector_type(8)))  bf16  bf16x8;
typedef __attribute__((ext_vector_type(16))) bf16  bf16x16;
typedef __attribute__((ext_vector_type(8)))  float f32x8;

static __device__ inline bf16x16 cat16(bf16x8 lo, bf16x8 hi) {
  union { bf16x16 v; bf16x8 h[2]; } u;
  u.h[0] = lo; u.h[1] = hi; return u.v;
}

// A fragment (16x32 bf16, row-major source, `stride` elements per row).
// ISA layout: lane = half*16+m ; elems 0..7 -> K=8*half+0..7 ; elems 8..15 -> K=8*half+16..23
static __device__ inline bf16x16 load_frag_a(const bf16* a, int stride, int k0, int lane) {
  int half = lane >> 4, m = lane & 15;
  const bf16* r = a + (size_t)m * stride + k0 + 8 * half;
  bf16x8 lo = *(const bf16x8*)(r);
  bf16x8 hi = *(const bf16x8*)(r + 16);
  return cat16(lo, hi);
}

// B fragment for C = A @ W^T with W[N][Kin] row-major: B[k][n] = W[n][k].
// lane = half*16+n ; elems 0..7 -> K = k0 + 8*half + 0..7 ; elems 8..15 -> K = k0 + 8*half + 16..23
static __device__ inline bf16x16 load_frag_b(const bf16* w, int kin, int n0, int k0, int lane) {
  int half = lane >> 4, n = lane & 15;
  const bf16* r = w + (size_t)(n0 + n) * kin + k0 + 8 * half;
  bf16x8 lo = *(const bf16x8*)(r);
  bf16x8 hi = *(const bf16x8*)(r + 16);
  return cat16(lo, hi);
}

#define WMMA_BF16(acc, a, b) \
  acc = __builtin_amdgcn_wmma_f32_16x16x32_bf16(false, (a), false, (b), (short)0, (acc), false, false)

static __device__ inline float geluf(float x) { return 0.5f * x * (1.0f + erff(x * 0.70710678118f)); }
static __device__ inline float siluf(float x) { return x / (1.0f + expf(-x)); }
static __device__ inline float sigmoidf_(float x) { return 1.0f / (1.0f + expf(-x)); }

// ---------------- kernel 1: f32 -> bf16 convert ----------------
__global__ void cvt_bf16_kernel(const float* __restrict__ src, bf16* __restrict__ dst, int n) {
  int i = blockIdx.x * blockDim.x + threadIdx.x;
  if (i < n) dst[i] = (bf16)src[i];
}

// ---------------- kernel 2: LayerNorm(feats)*g -> bf16 x ----------------
__global__ void ln_kernel(const float* __restrict__ feats, const float* __restrict__ g,
                          bf16* __restrict__ xo) {
  __shared__ float red[256];
  int i = blockIdx.x, t = threadIdx.x;
  float a = feats[(size_t)i * DD + t];
  float b = feats[(size_t)i * DD + 256 + t];
  red[t] = a + b; __syncthreads();
  for (int o = 128; o > 0; o >>= 1) { if (t < o) red[t] += red[t + o]; __syncthreads(); }
  float mean = red[0] * (1.0f / DD);
  __syncthreads();
  float da = a - mean, db = b - mean;
  red[t] = da * da + db * db; __syncthreads();
  for (int o = 128; o > 0; o >>= 1) { if (t < o) red[t] += red[t + o]; __syncthreads(); }
  float rstd = rsqrtf(red[0] * (1.0f / DD) + 1e-5f);
  xo[(size_t)i * DD + t]       = (bf16)(da * rstd * g[t]);
  xo[(size_t)i * DD + 256 + t] = (bf16)(db * rstd * g[256 + t]);
}

// ---------------- kernel 3: QKV GEMM (2048x512 @ 512x1536^T) via WMMA ----------------
// Register-blocked: each wave keeps 4 accumulator tiles live; one A frag feeds 4 B tiles.
// Writes q/k/v as [h][n][dh] f32.
__global__ void qkv_gemm_kernel(const bf16* __restrict__ x, const bf16* __restrict__ w,
                                float* __restrict__ qb, float* __restrict__ kb,
                                float* __restrict__ vb) {
  __shared__ bf16 at[16 * DD];  // 16KB A tile
  int t = threadIdx.x, lane = t & 31, wave = t >> 5;
  int m0 = blockIdx.x * 16;
  for (int e = t * 8; e < 16 * DD; e += 256 * 8)
    *(bf16x8*)(at + e) = *(const bf16x8*)(x + (size_t)m0 * DD + e);
  __syncthreads();
  int half = lane >> 4, nl = lane & 15;
  for (int g = 0; g < 3; ++g) {                 // 96 col tiles = 8 waves * 3 groups * 4
    int ctbase = wave * 12 + g * 4;
    f32x8 acc[4];
    #pragma unroll
    for (int j = 0; j < 4; ++j) acc[j] = (f32x8){0.f,0.f,0.f,0.f,0.f,0.f,0.f,0.f};
    #pragma unroll 2
    for (int k0 = 0; k0 < DD; k0 += 32) {
      bf16x16 a = load_frag_a(at, DD, k0, lane);
      #pragma unroll
      for (int j = 0; j < 4; ++j) {
        bf16x16 b = load_frag_b(w, DD, (ctbase + j) * 16, k0, lane);
        WMMA_BF16(acc[j], a, b);
      }
    }
    #pragma unroll
    for (int j = 0; j < 4; ++j) {
      int gn = (ctbase + j) * 16 + nl;
      int which = gn >> 9, rem = gn & 511, h = rem >> 6, dh = rem & 63;
      float* dst = (which == 0) ? qb : ((which == 1) ? kb : vb);
      #pragma unroll
      for (int r = 0; r < 8; r++) {
        int gm = m0 + r + 8 * half;
        dst[((size_t)h * NPT + gm) * DHD + dh] = acc[j][r];
      }
    }
  }
}

// ---------------- kernel 4: gates = sigmoid(x @ gw^T + gb) ----------------
__global__ void gates_kernel(const bf16* __restrict__ x, const float* __restrict__ gw,
                             const float* __restrict__ gb, float* __restrict__ gout) {
  __shared__ float red[256];
  int i = blockIdx.x, t = threadIdx.x;
  int g = t & 15, chunk = t >> 4;
  float s = 0.f;
  for (int c = chunk * 32; c < chunk * 32 + 32; c++)
    s += (float)x[(size_t)i * DD + c] * gw[(size_t)g * DD + c];
  red[t] = s; __syncthreads();
  if (t < 16) {
    float tot = 0.f;
    for (int c = 0; c < 16; c++) tot += red[t + c * 16];
    gout[(size_t)i * 16 + t] = sigmoidf_(tot + gb[t]);
  }
}

// ---------------- kernel 5: top-K nearest neighbors ----------------
__global__ void topk_kernel(const float* __restrict__ coors, int* __restrict__ idxb,
                            float* __restrict__ sqb) {
  __shared__ float dist[NPT];
  __shared__ float rv[256];
  __shared__ int   ri[256];
  int i = blockIdx.x, t = threadIdx.x;
  float cx = coors[i * 3], cy = coors[i * 3 + 1], cz = coors[i * 3 + 2];
  for (int j = t; j < NPT; j += 256) {
    float dx = cx - coors[j * 3], dy = cy - coors[j * 3 + 1], dz = cz - coors[j * 3 + 2];
    dist[j] = dx * dx + dy * dy + dz * dz;
  }
  __syncthreads();
  for (int kk = 0; kk < KNN; kk++) {
    float bv = INFINITY; int bi = NPT;
    for (int j = t; j < NPT; j += 256) {
      float v = dist[j];
      if (v < bv) { bv = v; bi = j; }
    }
    rv[t] = bv; ri[t] = bi; __syncthreads();
    for (int o = 128; o > 0; o >>= 1) {
      if (t < o) {
        float v2 = rv[t + o]; int i2 = ri[t + o];
        if (v2 < rv[t] || (v2 == rv[t] && i2 < ri[t])) { rv[t] = v2; ri[t] = i2; }
      }
      __syncthreads();
    }
    if (t == 0) { idxb[i * KNN + kk] = ri[0]; sqb[i * KNN + kk] = rv[0]; dist[ri[0]] = INFINITY; }
    __syncthreads();
  }
}

// ---------------- LN+SiLU over a 32x256 LDS tile ----------------
static __device__ void ln_silu_32x256(float* fb, bf16* p, const float* g, const float* b,
                                      float* red1, float* red2, float* mrow, float* rrow, int t) {
  int row = t >> 3, sub = t & 7;
  float s = 0.f, ss = 0.f;
  for (int c = sub * 32; c < sub * 32 + 32; c++) { float v = fb[row * PBD + c]; s += v; ss += v * v; }
  red1[t] = s; red2[t] = ss; __syncthreads();
  if (t < KNN) {
    float S = 0.f, SS = 0.f;
    for (int q = 0; q < 8; q++) { S += red1[t * 8 + q]; SS += red2[t * 8 + q]; }
    float m = S * (1.0f / PBD);
    mrow[t] = m;
    rrow[t] = rsqrtf(SS * (1.0f / PBD) - m * m + 1e-5f);
  }
  __syncthreads();
  for (int r = 0; r < KNN; r++) {
    float v = (fb[r * PBD + t] - mrow[r]) * rrow[r] * g[t] + b[t];
    p[r * PBD + t] = (bf16)siluf(v);
  }
  __syncthreads();
}

// ---------------- kernel 6: fused per-node pipeline ----------------
// One block per node i. DPB MLP (WMMA), vpos+v gather (WMMA), qk, edge MLP,
// both softmaxes, attention output, coordinate output. All intermediates in LDS.
#define MEGA_SMEM_BYTES (16384 + 32768 + 65536 + 5208 * 4)

__global__ void mega_kernel(
    const float* __restrict__ coors, const float* __restrict__ edges,
    const float* __restrict__ qb, const float* __restrict__ kb, const float* __restrict__ vb,
    const int* __restrict__ idxb, const float* __restrict__ sqb, const float* __restrict__ gatesb,
    const float* __restrict__ w0, const float* __restrict__ b0,
    const float* __restrict__ ln0g, const float* __restrict__ ln0b,
    const bf16* __restrict__ whb, const float* __restrict__ bh,
    const float* __restrict__ lnhg, const float* __restrict__ lnhb,
    const float* __restrict__ wqp, const float* __restrict__ bqp,
    const bf16* __restrict__ wvpb, const float* __restrict__ bvp,
    const float* __restrict__ we1, const float* __restrict__ we2,
    const float* __restrict__ wc, const float* __restrict__ wcg,
    const float* __restrict__ bcg, const float* __restrict__ cns,
    const float* __restrict__ ccomb,
    bf16* __restrict__ outb, float* __restrict__ coors_out) {
  extern __shared__ char smem[];
  bf16*  sp  = (bf16*)smem;                    // 32x256 p (bf16)
  float* fb  = (float*)(smem + 16 * 1024);     // 32x256 pre-act
  float* sv2 = (float*)(smem + 48 * 1024);     // 32x512 v2
  float* f   = (float*)(smem + 112 * 1024);
  float* red1 = f; f += 256;
  float* red2 = f; f += 256;
  float* mrow = f; f += 32;
  float* rrow = f; f += 32;
  int*   sidx = (int*)f; f += 32;
  float* ssq  = f; f += 32;
  float* src_ = f; f += 96;   // rel coors 32x3
  float* qc   = f; f += 512;  // q[i] cache
  float* sqkp = f; f += 256;  // qk_pos 32x8
  float* sqk1 = f; f += 256;  // pre-MLP qk 32x8
  float* seg  = f; f += 512;  // gathered edges 32x16
  float* shid = f; f += 1536; // edge-MLP hidden 32x48
  float* sqk2 = f; f += 256;  // post-MLP qk_t 32x8
  float* sattn= f; f += 256;  // softmax attn 8x32
  float* scg  = f; f += 256;  // coor logits 32x8
  float* sca  = f; f += 256;  // coor attn 8x32
  float* ssgn = f; f += 256;  // sign 32x8
  float* scn  = f; f += 96;   // unit rel coors 32x3
  float* sro  = f; f += 24;   // rel_out 8x3

  int i = blockIdx.x, t = threadIdx.x, lane = t & 31, wave = t >> 5;
  int half = lane >> 4, nl = lane & 15;

  if (t < KNN) { sidx[t] = idxb[i * KNN + t]; ssq[t] = sqb[i * KNN + t]; }
  __syncthreads();
  if (t < KNN * 3) {
    int k = t / 3, c = t % 3;
    src_[t] = coors[i * 3 + c] - coors[sidx[k] * 3 + c];
  }
  for (int n = t; n < NH * DHD; n += 256)
    qc[n] = qb[((size_t)(n >> 6) * NPT + i) * DHD + (n & 63)];
  for (int e = t; e < KNN * DEE; e += 256) {
    int k = e >> 4, c = e & 15;
    seg[e] = edges[((size_t)i * NPT + sidx[k]) * DEE + c];
  }
  // DPB layer 0: fb[r][c] = (-sq[r]) * w0[c] + b0[c]
  {
    float w0c = w0[t], b0c = b0[t];
    for (int r = 0; r < KNN; r++) fb[r * PBD + t] = (-ssq[r]) * w0c + b0c;
  }
  __syncthreads();
  ln_silu_32x256(fb, sp, ln0g, ln0b, red1, red2, mrow, rrow, t);

  // two hidden layers 256->256 (WMMA) with bias + LN + SiLU
  // wave -> row tile m0 = (wave&1)*16 ; 4 consecutive N tiles nbase = (wave>>1)*4
  for (int l = 0; l < 2; l++) {
    const bf16* W = whb + (size_t)l * PBD * PBD;
    int m0 = (wave & 1) * 16;
    int nbase = (wave >> 1) * 4;
    f32x8 acc[4];
    #pragma unroll
    for (int j = 0; j < 4; ++j) acc[j] = (f32x8){0.f,0.f,0.f,0.f,0.f,0.f,0.f,0.f};
    #pragma unroll 2
    for (int k0 = 0; k0 < PBD; k0 += 32) {
      bf16x16 a = load_frag_a(sp + m0 * PBD, PBD, k0, lane);
      #pragma unroll
      for (int j = 0; j < 4; ++j) {
        bf16x16 b = load_frag_b(W, PBD, (nbase + j) * 16, k0, lane);
        WMMA_BF16(acc[j], a, b);
      }
    }
    #pragma unroll
    for (int j = 0; j < 4; ++j) {
      int gn = (nbase + j) * 16 + nl;
      float bias = bh[l * PBD + gn];
      #pragma unroll
      for (int r = 0; r < 8; r++) fb[(m0 + r + 8 * half) * PBD + gn] = acc[j][r] + bias;
    }
    __syncthreads();
    ln_silu_32x256(fb, sp, lnhg + l * PBD, lnhb + l * PBD, red1, red2, mrow, rrow, t);
  }

  // qk_pos = p @ wq_pos^T + bq_pos (32x8, scalar)
  {
    int r = t >> 3, h = t & 7;
    float s = bqp[h];
    for (int c = 0; c < PBD; c++) s += (float)sp[r * PBD + c] * wqp[(size_t)h * PBD + c];
    sqkp[r * NH + h] = s;
  }
  // v2 = p @ wv_pos^T + bv_pos + gather(v)  (32x512, WMMA)
  // wave -> m0 = (wave&1)*16 ; 8 N tiles in 2 groups of 4: nb = (wave>>1)*8
  {
    int m0 = (wave & 1) * 16;
    int nb = (wave >> 1) * 8;
    for (int g = 0; g < 2; ++g) {
      int nbase = nb + g * 4;
      f32x8 acc[4];
      #pragma unroll
      for (int j = 0; j < 4; ++j) acc[j] = (f32x8){0.f,0.f,0.f,0.f,0.f,0.f,0.f,0.f};
      #pragma unroll 2
      for (int k0 = 0; k0 < PBD; k0 += 32) {
        bf16x16 a = load_frag_a(sp + m0 * PBD, PBD, k0, lane);
        #pragma unroll
        for (int j = 0; j < 4; ++j) {
          bf16x16 b = load_frag_b(wvpb, PBD, (nbase + j) * 16, k0, lane);
          WMMA_BF16(acc[j], a, b);
        }
      }
      #pragma unroll
      for (int j = 0; j < 4; ++j) {
        int gn = (nbase + j) * 16 + nl;
        int h = gn >> 6, dh = gn & 63;
        float bias = bvp[gn];
        #pragma unroll
        for (int r = 0; r < 8; r++) {
          int kk = m0 + r + 8 * half;
          int jj = sidx[kk];
          sv2[kk * 512 + gn] = acc[j][r] + bias + vb[((size_t)h * NPT + jj) * DHD + dh];
        }
      }
    }
  }
  __syncthreads();

  // qk = -||q - k_j||^2 * scale + qk_pos   (8x32)
  {
    int h = t >> 5, k = t & 31;
    int j = sidx[k];
    const float* kp = kb + ((size_t)h * NPT + j) * DHD;
    float s = 0.f;
    for (int dh = 0; dh < DHD; dh++) { float dv = qc[h * DHD + dh] - kp[dh]; s += dv * dv; }
    sqk1[k * NH + h] = -s * ATT_SCALE + sqkp[k * NH + h];
  }
  __syncthreads();
  // edge MLP: gelu([qk_t, eg] @ we1^T) @ we2^T
  {
    int k = t & 31, u0 = t >> 5;
    for (int jj = 0; jj < 6; jj++) {
      int u = u0 + 8 * jj;
      const float* wr = we1 + u * 24;
      float s = 0.f;
      for (int h = 0; h < NH; h++)  s += sqk1[k * NH + h] * wr[h];
      for (int c = 0; c < DEE; c++) s += seg[k * DEE + c] * wr[NH + c];
      shid[k * 48 + u] = geluf(s);
    }
  }
  __syncthreads();
  {
    int k = t & 31, h = t >> 5;
    float s = 0.f;
    for (int u = 0; u < 48; u++) s += shid[k * 48 + u] * we2[h * 48 + u];
    sqk2[k * NH + h] = s;
  }
  __syncthreads();
  // coordinate logits + sign + unit rel coors
  {
    int k = t & 31, h = t >> 5;
    float a = 0.f, b = 0.f;
    for (int h2 = 0; h2 < NH; h2++) {
      float q2 = sqk2[k * NH + h2];
      a += geluf(q2) * wc[h * NH + h2];
      b += q2 * wcg[h * NH + h2];
    }
    scg[k * NH + h]  = a;
    ssgn[k * NH + h] = b + bcg[h];
  }
  if (t < KNN) {
    float x = src_[t * 3], y = src_[t * 3 + 1], z = src_[t * 3 + 2];
    float inv = rsqrtf(fmaxf(x * x + y * y + z * z, 1e-16f)) * cns[0];
    scn[t * 3] = x * inv; scn[t * 3 + 1] = y * inv; scn[t * 3 + 2] = z * inv;
  }
  __syncthreads();
  // softmaxes over k (per head)
  if (t < NH) {
    float mx = -INFINITY;
    for (int k = 0; k < KNN; k++) mx = fmaxf(mx, sqk2[k * NH + t]);
    float sum = 0.f;
    for (int k = 0; k < KNN; k++) { float e = expf(sqk2[k * NH + t] - mx); sattn[t * 32 + k] = e; sum += e; }
    float inv = 1.0f / sum;
    for (int k = 0; k < KNN; k++) sattn[t * 32 + k] *= inv;
    mx = -INFINITY;
    for (int k = 0; k < KNN; k++) mx = fmaxf(mx, scg[k * NH + t]);
    sum = 0.f;
    for (int k = 0; k < KNN; k++) { float e = expf(scg[k * NH + t] - mx); sca[t * 32 + k] = e; sum += e; }
    inv = 1.0f / sum;
    for (int k = 0; k < KNN; k++) sca[t * 32 + k] *= inv;
  }
  __syncthreads();
  // coordinate output
  if (t < 24) {
    int h = t / 3, c = t % 3;
    float s = 0.f;
    for (int k = 0; k < KNN; k++) s += sca[h * 32 + k] * scn[k * 3 + c] * ssgn[k * NH + h];
    sro[t] = s * gatesb[(size_t)i * 16 + 8 + h];
  }
  __syncthreads();
  if (t < 3) {
    float s = 0.f;
    for (int h = 0; h < NH; h++) s += sro[h * 3 + t] * ccomb[h];
    coors_out[(size_t)i * 3 + t] = s;
  }
  // attention output -> bf16 row for final GEMM
  for (int n = t; n < 512; n += 256) {
    int h = n >> 6;
    float s = 0.f;
    for (int k = 0; k < KNN; k++) s += sattn[h * 32 + k] * sv2[k * 512 + n];
    outb[(size_t)i * 512 + n] = (bf16)(s * gatesb[(size_t)i * 16 + h]);
  }
}

// ---------------- kernel 7: node_out = out @ w_out^T + b_out (WMMA) ----------------
__global__ void out_gemm_kernel(const bf16* __restrict__ a, const bf16* __restrict__ w,
                                const float* __restrict__ bias, float* __restrict__ outp) {
  __shared__ bf16 at[16 * DD];
  int t = threadIdx.x, lane = t & 31, wave = t >> 5;
  int m0 = blockIdx.x * 16;
  for (int e = t * 8; e < 16 * DD; e += 256 * 8)
    *(bf16x8*)(at + e) = *(const bf16x8*)(a + (size_t)m0 * DD + e);
  __syncthreads();
  int half = lane >> 4, nl = lane & 15;
  int ctbase = wave * 4;                        // 32 col tiles = 8 waves * 4
  f32x8 acc[4];
  #pragma unroll
  for (int j = 0; j < 4; ++j) acc[j] = (f32x8){0.f,0.f,0.f,0.f,0.f,0.f,0.f,0.f};
  #pragma unroll 2
  for (int k0 = 0; k0 < DD; k0 += 32) {
    bf16x16 af = load_frag_a(at, DD, k0, lane);
    #pragma unroll
    for (int j = 0; j < 4; ++j) {
      bf16x16 bfg = load_frag_b(w, DD, (ctbase + j) * 16, k0, lane);
      WMMA_BF16(acc[j], af, bfg);
    }
  }
  #pragma unroll
  for (int j = 0; j < 4; ++j) {
    int gn = (ctbase + j) * 16 + nl;
    float bi = bias[gn];
    #pragma unroll
    for (int r = 0; r < 8; r++) {
      int gm = m0 + r + 8 * half;
      outp[(size_t)gm * DD + gn] = acc[j][r] + bi;
    }
  }
}

// ---------------- host launcher ----------------
extern "C" void kernel_launch(void* const* d_in, const int* in_sizes, int n_in,
                              void* d_out, int out_size, void* d_ws, size_t ws_size,
                              hipStream_t stream) {
  (void)in_sizes; (void)n_in; (void)out_size; (void)ws_size;
  const float* feats   = (const float*)d_in[0];
  const float* coors   = (const float*)d_in[1];
  const float* edges   = (const float*)d_in[2];
  const float* ln_g    = (const float*)d_in[3];
  const float* w_qkv   = (const float*)d_in[4];
  const float* w_out   = (const float*)d_in[5];
  const float* b_out   = (const float*)d_in[6];
  const float* gate_w  = (const float*)d_in[7];
  const float* gate_b  = (const float*)d_in[8];
  const float* we1     = (const float*)d_in[9];
  const float* we2     = (const float*)d_in[10];
  const float* wc      = (const float*)d_in[11];
  const float* wcg     = (const float*)d_in[12];
  const float* bcg     = (const float*)d_in[13];
  const float* cns     = (const float*)d_in[14];
  const float* ccomb   = (const float*)d_in[15];
  const float* dpb_w0  = (const float*)d_in[16];
  const float* dpb_b0  = (const float*)d_in[17];
  const float* ln0g    = (const float*)d_in[18];
  const float* ln0b    = (const float*)d_in[19];
  const float* dpb_wh  = (const float*)d_in[20];
  const float* dpb_bh  = (const float*)d_in[21];
  const float* lnhg    = (const float*)d_in[22];
  const float* lnhb    = (const float*)d_in[23];
  const float* wq_pos  = (const float*)d_in[24];
  const float* bq_pos  = (const float*)d_in[25];
  const float* wv_pos  = (const float*)d_in[26];
  const float* bv_pos  = (const float*)d_in[27];

  char* ws = (char*)d_ws;
  auto alloc = [&](size_t bytes) -> char* {
    char* p = ws; ws += (bytes + 255) & ~(size_t)255; return p;
  };
  bf16*  x_bf    = (bf16*)alloc((size_t)NPT * DD * 2);
  bf16*  wqkv_bf = (bf16*)alloc((size_t)3 * NH * DHD * DD * 2);
  bf16*  wout_bf = (bf16*)alloc((size_t)DD * NH * DHD * 2);
  bf16*  wh_bf   = (bf16*)alloc((size_t)2 * PBD * PBD * 2);
  bf16*  wvp_bf  = (bf16*)alloc((size_t)NH * DHD * PBD * 2);
  float* qb      = (float*)alloc((size_t)NH * NPT * DHD * 4);
  float* kb      = (float*)alloc((size_t)NH * NPT * DHD * 4);
  float* vb      = (float*)alloc((size_t)NH * NPT * DHD * 4);
  int*   idxb    = (int*)  alloc((size_t)NPT * KNN * 4);
  float* sqb     = (float*)alloc((size_t)NPT * KNN * 4);
  float* gatesb  = (float*)alloc((size_t)NPT * 16 * 4);
  bf16*  outb    = (bf16*)alloc((size_t)NPT * DD * 2);

  float* node_out  = (float*)d_out;
  float* coors_out = (float*)d_out + (size_t)NPT * DD;

  // weight conversions
  cvt_bf16_kernel<<<(3 * NH * DHD * DD + 255) / 256, 256, 0, stream>>>(w_qkv, wqkv_bf, 3 * NH * DHD * DD);
  cvt_bf16_kernel<<<(DD * NH * DHD + 255) / 256, 256, 0, stream>>>(w_out, wout_bf, DD * NH * DHD);
  cvt_bf16_kernel<<<(2 * PBD * PBD + 255) / 256, 256, 0, stream>>>(dpb_wh, wh_bf, 2 * PBD * PBD);
  cvt_bf16_kernel<<<(NH * DHD * PBD + 255) / 256, 256, 0, stream>>>(wv_pos, wvp_bf, NH * DHD * PBD);

  ln_kernel<<<NPT, 256, 0, stream>>>(feats, ln_g, x_bf);
  qkv_gemm_kernel<<<NPT / 16, 256, 0, stream>>>(x_bf, wqkv_bf, qb, kb, vb);
  gates_kernel<<<NPT, 256, 0, stream>>>(x_bf, gate_w, gate_b, gatesb);
  topk_kernel<<<NPT, 256, 0, stream>>>(coors, idxb, sqb);

  (void)hipFuncSetAttribute((const void*)mega_kernel,
                            hipFuncAttributeMaxDynamicSharedMemorySize, MEGA_SMEM_BYTES);
  mega_kernel<<<NPT, 256, MEGA_SMEM_BYTES, stream>>>(
      coors, edges, qb, kb, vb, idxb, sqb, gatesb,
      dpb_w0, dpb_b0, ln0g, ln0b, wh_bf, dpb_bh, lnhg, lnhb,
      wq_pos, bq_pos, wvp_bf, bv_pos, we1, we2, wc, wcg, bcg, cns, ccomb,
      outb, coors_out);

  out_gemm_kernel<<<NPT / 16, 256, 0, stream>>>(outb, wout_bf, b_out, node_out);
}
